// CrystalGraphALIGNN_30107720745192
// MI455X (gfx1250) — matrix operand order repaired and checked by reading, not compiled, not run on hardware
//
#include <hip/hip_runtime.h>
#include <stdint.h>

// ---------------- problem constants ----------------
#define NNODE   50000
#define MNBR    12
#define NEDGE   (NNODE * MNBR)      // 600000
#define AF      92
#define EF      41
#define ND      128
#define ED      64
#define HID     128
#define RD      128
#define NLAYER  4
#define NCRYS   1000

// ---------------- WMMA types / helpers ----------------
typedef __attribute__((ext_vector_type(16))) __bf16 v16bf;
typedef __attribute__((ext_vector_type(8)))  float  v8f;

__device__ inline unsigned short f2bf(float x) {
    unsigned int u = __builtin_bit_cast(unsigned int, x);
    unsigned int r = u + 0x7fffu + ((u >> 16) & 1u);   // round-to-nearest-even
    return (unsigned short)(r >> 16);
}

__device__ inline float silu(float x) {
    return x * (1.0f / (1.0f + __expf(-x)));
}

// A fragment (16x32 bf16) from an LDS row. rowBase = &lds[row][kc*32].
// k order used here (contiguous, same as prepack B): lanes 0-15 -> elems {K0..15}; lanes 16-31 -> {K16..31}
__device__ inline v16bf load_a_frag(const unsigned short* rowBase, int lane) {
    int kk = ((lane >> 4) << 4);           // 0 or 16 elements
    v16bf a;
    ((float4*)&a)[0] = *(const float4*)(rowBase + kk);
    ((float4*)&a)[1] = *(const float4*)(rowBase + kk + 8);
    return a;
}

// B fragment from prepacked global buffer (32 bytes contiguous per lane)
__device__ inline v16bf load_b_frag(const unsigned short* p) {
    v16bf b;
    ((float4*)&b)[0] = ((const float4*)p)[0];
    ((float4*)&b)[1] = ((const float4*)p)[1];
    return b;
}

__device__ inline v8f wmma_bf16(v16bf a, v16bf b, v8f c) {
    return __builtin_amdgcn_wmma_f32_16x16x32_bf16(false, a, false, b, (short)0, c,
                                                   false, false);
}

// ---------------- weight prepack: fp32 [K][Ncols] -> bf16 B-fragment layout ----------------
// packed[(kc*NT + nt)*32 + lane][j] = W[kc*32 + (lane>>4)*16 + j][nt*16 + (lane&15)]
__global__ __launch_bounds__(32)
void k_prepack(const float* __restrict__ src, unsigned short* __restrict__ dst,
               int K, int Ncols, int srcLayerStride, int dstLayerStride) {
    int kc = blockIdx.x, nt = blockIdx.y, l = blockIdx.z;
    int NT = Ncols >> 4;
    int lane = threadIdx.x;
    const float* s = src + (size_t)l * srcLayerStride;
    unsigned short* d = dst + (size_t)l * dstLayerStride +
                        (((size_t)(kc * NT + nt)) * 32 + lane) * 16;
    int kbase = kc * 32 + ((lane >> 4) << 4);
    int n = nt * 16 + (lane & 15);
#pragma unroll
    for (int j = 0; j < 16; ++j) {
        int k = kbase + j;
        float v = (k < K) ? s[(size_t)k * Ncols + n] : 0.0f;
        d[j] = f2bf(v);
    }
}

__global__ __launch_bounds__(256)
void k_zero(float* __restrict__ p, int n) {
    int t = blockIdx.x * 256 + threadIdx.x;
    if (t < n) p[t] = 0.0f;
}

// ---------------- node projection: [N,92] @ [92,128] + b ----------------
__global__ __launch_bounds__(256)
void k_node_proj(const float* __restrict__ A, const unsigned short* __restrict__ pW,
                 const float* __restrict__ bias, float* __restrict__ out) {
    __shared__ alignas(16) unsigned short sA[16][96];   // K padded 92->96
    int tid = threadIdx.x;
    int r0 = blockIdx.x * 16;
    for (int q = tid; q < 16 * 96; q += 256) {
        int r = q / 96, f = q - r * 96;
        float v = (f < AF) ? A[(size_t)(r0 + r) * AF + f] : 0.0f;
        sA[r][f] = f2bf(v);
    }
    __syncthreads();
    int lane = tid & 31, wave = tid >> 5;               // wave = column tile 0..7
    const unsigned short* aRow = sA[lane & 15];
    v8f acc = {};
#pragma unroll
    for (int kc = 0; kc < 3; ++kc) {
        v16bf a = load_a_frag(aRow + kc * 32, lane);
        v16bf b = load_b_frag(pW + (((kc * 8 + wave) * 32 + lane) * 16));
        acc = wmma_bf16(a, b, acc);
    }
    int col = lane & 15, rb = (lane >> 4) * 8;
    float bv = bias[wave * 16 + col];
#pragma unroll
    for (int v = 0; v < 8; ++v)
        out[(size_t)(r0 + rb + v) * ND + wave * 16 + col] = acc[v] + bv;
}

// ---------------- edge projection: [E,41] @ [41,64] + b, plus edge mask ----------------
__global__ __launch_bounds__(128)
void k_edge_proj(const float* __restrict__ attr, const unsigned short* __restrict__ pW,
                 const float* __restrict__ bias, float* __restrict__ edge,
                 float* __restrict__ maskb) {
    __shared__ alignas(16) unsigned short sA[16][64];   // K padded 41->64
    int tid = threadIdx.x;
    int e0 = blockIdx.x * 16;
    for (int q = tid; q < 16 * 64; q += 128) {
        int r = q >> 6, f = q & 63;
        float v = (f < EF) ? attr[(size_t)(e0 + r) * EF + f] : 0.0f;
        sA[r][f] = f2bf(v);
    }
    if (tid < 16) {
        int e = e0 + tid;
        float s = 0.0f;
        for (int f = 0; f < EF; ++f) s += fabsf(attr[(size_t)e * EF + f]);
        maskb[e] = (s > 1e-6f) ? 1.0f : 0.0f;
    }
    __syncthreads();
    int lane = tid & 31, wave = tid >> 5;               // 4 waves -> 4 col tiles of 64
    const unsigned short* aRow = sA[lane & 15];
    v8f acc = {};
#pragma unroll
    for (int kc = 0; kc < 2; ++kc) {
        v16bf a = load_a_frag(aRow + kc * 32, lane);
        v16bf b = load_b_frag(pW + (((kc * 4 + wave) * 32 + lane) * 16));
        acc = wmma_bf16(a, b, acc);
    }
    int col = lane & 15, rb = (lane >> 4) * 8;
    float bv = bias[wave * 16 + col];
#pragma unroll
    for (int v = 0; v < 8; ++v)
        edge[(size_t)(e0 + rb + v) * ED + wave * 16 + col] = acc[v] + bv;
}

// ---------------- fused edge update + masked scatter-add (32-edge tile) ----------------
// edge += SiLU([edge|node[src]|node[dst]] @ W1 + b1) @ W2 + b2 ; atomic scatter into agg
__global__ __launch_bounds__(256)
void k_edge_update(const float* __restrict__ node, float* __restrict__ edge,
                   const float* __restrict__ maskb, const int* __restrict__ nbr,
                   const unsigned short* __restrict__ pW1, const float* __restrict__ b1,
                   const unsigned short* __restrict__ pW2, const float* __restrict__ b2,
                   float* __restrict__ aggS, float* __restrict__ aggC) {
    __shared__ alignas(16) unsigned short sA[32][320];  // concat panel, bf16 (20KB)
    __shared__ alignas(16) unsigned short sH[32][128];  // hidden, bf16 (8KB)
    __shared__ float sE[32][64];                        // fp32 edge rows (8KB)
    __shared__ int   sDst[32];
    __shared__ float sMask[32];
    int tid = threadIdx.x;
    int e0 = blockIdx.x * 32;

    // stage: edge features K[0,64) (fp32 copy kept for residual)
    for (int q = tid; q < 32 * 64; q += 256) {
        int r = q >> 6, f = q & 63;
        float v = edge[(size_t)(e0 + r) * ED + f];
        sE[r][f] = v;
        sA[r][f] = f2bf(v);
    }
    // stage: node[src] K[64,192)
    for (int q = tid; q < 32 * 128; q += 256) {
        int r = q >> 7, f = q & 127;
        int s = (e0 + r) / MNBR;
        sA[r][64 + f] = f2bf(node[(size_t)s * ND + f]);
    }
    // stage: node[dst] K[192,320)
    for (int q = tid; q < 32 * 128; q += 256) {
        int r = q >> 7, f = q & 127;
        int d = nbr[e0 + r];
        d = d < 0 ? 0 : (d >= NNODE ? NNODE - 1 : d);
        sA[r][192 + f] = f2bf(node[(size_t)d * ND + f]);
    }
    if (tid < 32) {
        int e = e0 + tid;
        int d = nbr[e];
        d = d < 0 ? 0 : (d >= NNODE ? NNODE - 1 : d);
        sDst[tid] = d;
        sMask[tid] = maskb[e];
    }
    __syncthreads();

    int lane = tid & 31, wave = tid >> 5;
    int col = lane & 15, rb = (lane >> 4) * 8;

    // GEMM1: K=320 (10 chunks), 8 waves x 16 cols; 2 row-tiles share each B fragment
    const unsigned short* aRow0 = sA[lane & 15];
    const unsigned short* aRow1 = sA[16 + (lane & 15)];
    v8f acc0 = {}, acc1 = {};
#pragma unroll
    for (int kc = 0; kc < 10; ++kc) {
        v16bf b = load_b_frag(pW1 + (((kc * 8 + wave) * 32 + lane) * 16));
        if (kc + 1 < 10)
            __builtin_prefetch(pW1 + ((((kc + 1) * 8 + wave) * 32 + lane) * 16), 0, 3);
        v16bf a0 = load_a_frag(aRow0 + kc * 32, lane);
        v16bf a1 = load_a_frag(aRow1 + kc * 32, lane);
        acc0 = wmma_bf16(a0, b, acc0);
        acc1 = wmma_bf16(a1, b, acc1);
    }
    {
        float bv = b1[wave * 16 + col];
#pragma unroll
        for (int v = 0; v < 8; ++v) {
            sH[rb + v][wave * 16 + col]      = f2bf(silu(acc0[v] + bv));
            sH[16 + rb + v][wave * 16 + col] = f2bf(silu(acc1[v] + bv));
        }
    }
    __syncthreads();

    // GEMM2: K=128 (4 chunks); 8 waves = 2 row-tiles x 4 col-tiles; residual write
    {
        int rt = wave >> 2, nt2 = wave & 3;
        const unsigned short* hRow = sH[rt * 16 + (lane & 15)];
        v8f acc2 = {};
#pragma unroll
        for (int kc = 0; kc < 4; ++kc) {
            v16bf a = load_a_frag(hRow + kc * 32, lane);
            v16bf b = load_b_frag(pW2 + (((kc * 4 + nt2) * 32 + lane) * 16));
            acc2 = wmma_bf16(a, b, acc2);
        }
        float bv = b2[nt2 * 16 + col];
#pragma unroll
        for (int v = 0; v < 8; ++v) {
            int r = rt * 16 + rb + v, f = nt2 * 16 + col;
            float nv = sE[r][f] + acc2[v] + bv;
            sE[r][f] = nv;
            edge[(size_t)(e0 + r) * ED + f] = nv;
        }
    }
    __syncthreads();

    // masked scatter-add of new edge into agg sums / counts
    for (int q = tid; q < 32 * 64; q += 256) {
        int r = q >> 6, f = q & 63;
        float mk = sMask[r];
        if (mk != 0.0f)
            atomicAdd(&aggS[(size_t)sDst[r] * ED + f], sE[r][f] * mk);
    }
    if (tid < 32 && sMask[tid] != 0.0f) atomicAdd(&aggC[sDst[tid]], sMask[tid]);
}

// ---------------- fused node update ----------------
// node += SiLU([node|agg] @ W1 + b1) @ W2 + b2
__global__ __launch_bounds__(256)
void k_node_update(float* __restrict__ node, const float* __restrict__ aggS,
                   const float* __restrict__ aggC,
                   const unsigned short* __restrict__ pW1, const float* __restrict__ b1,
                   const unsigned short* __restrict__ pW2, const float* __restrict__ b2) {
    __shared__ alignas(16) unsigned short sA[16][192];
    __shared__ alignas(16) unsigned short sH[16][128];
    int tid = threadIdx.x;
    int i0 = blockIdx.x * 16;

    for (int q = tid; q < 16 * 128; q += 256) {
        int r = q >> 7, f = q & 127;
        sA[r][f] = f2bf(node[(size_t)(i0 + r) * ND + f]);
    }
    for (int q = tid; q < 16 * 64; q += 256) {
        int r = q >> 6, f = q & 63;
        int i = i0 + r;
        float den = fmaxf(aggC[i], 1.0f);
        sA[r][128 + f] = f2bf(aggS[(size_t)i * ED + f] / den);
    }
    __syncthreads();

    int lane = tid & 31, wave = tid >> 5;
    int col = lane & 15, rb = (lane >> 4) * 8;
    const unsigned short* aRow = sA[lane & 15];

    v8f acc = {};
#pragma unroll
    for (int kc = 0; kc < 6; ++kc) {
        v16bf a = load_a_frag(aRow + kc * 32, lane);
        v16bf b = load_b_frag(pW1 + (((kc * 8 + wave) * 32 + lane) * 16));
        acc = wmma_bf16(a, b, acc);
    }
    {
        float bv = b1[wave * 16 + col];
#pragma unroll
        for (int v = 0; v < 8; ++v)
            sH[rb + v][wave * 16 + col] = f2bf(silu(acc[v] + bv));
    }
    __syncthreads();

    const unsigned short* hRow = sH[lane & 15];
    v8f acc2 = {};
#pragma unroll
    for (int kc = 0; kc < 4; ++kc) {
        v16bf a = load_a_frag(hRow + kc * 32, lane);
        v16bf b = load_b_frag(pW2 + (((kc * 8 + wave) * 32 + lane) * 16));
        acc2 = wmma_bf16(a, b, acc2);
    }
    float bv = b2[wave * 16 + col];
#pragma unroll
    for (int v = 0; v < 8; ++v) {
        size_t gi = (size_t)(i0 + rb + v) * ND + wave * 16 + col;
        node[gi] = node[gi] + acc2[v] + bv;
    }
}

// ---------------- per-crystal pooling (atomic) ----------------
__global__ __launch_bounds__(256)
void k_pool(const float* __restrict__ node, const int* __restrict__ cb,
            float* __restrict__ pS, float* __restrict__ pC) {
    long t = (long)blockIdx.x * 256 + threadIdx.x;
    if (t < (long)NNODE * ND) {
        int i = (int)(t >> 7), f = (int)(t & 127);
        int c = cb[i];
        c = c < 0 ? 0 : (c >= NCRYS ? NCRYS - 1 : c);
        atomicAdd(&pS[(size_t)c * ND + f], node[t]);
        if (f == 0) atomicAdd(&pC[c], 1.0f);
    }
}

// ---------------- readout head: SiLU(pooled @ rW + rb) @ oW + ob ----------------
__global__ __launch_bounds__(256)
void k_readout(const float* __restrict__ poolS, const float* __restrict__ poolC,
               const unsigned short* __restrict__ pRW, const float* __restrict__ rbv,
               const float* __restrict__ oW, const float* __restrict__ ob,
               float* __restrict__ out) {
    __shared__ alignas(16) unsigned short sA[16][128];
    __shared__ float sC[16][128];
    int tid = threadIdx.x;
    int c0 = blockIdx.x * 16;
    for (int q = tid; q < 16 * 128; q += 256) {
        int r = q >> 7, f = q & 127;
        int c = c0 + r;
        float v = 0.0f;
        if (c < NCRYS) {
            float den = fmaxf(poolC[c], 1.0f);
            v = poolS[(size_t)c * ND + f] / den;
        }
        sA[r][f] = f2bf(v);
    }
    __syncthreads();
    int lane = tid & 31, wave = tid >> 5;
    const unsigned short* aRow = sA[lane & 15];
    v8f acc = {};
#pragma unroll
    for (int kc = 0; kc < 4; ++kc) {
        v16bf a = load_a_frag(aRow + kc * 32, lane);
        v16bf b = load_b_frag(pRW + (((kc * 8 + wave) * 32 + lane) * 16));
        acc = wmma_bf16(a, b, acc);
    }
    int col = lane & 15, rb = (lane >> 4) * 8;
    float bv = rbv[wave * 16 + col];
#pragma unroll
    for (int v = 0; v < 8; ++v)
        sC[rb + v][wave * 16 + col] = silu(acc[v] + bv);
    __syncthreads();
    if (tid < 16) {
        int c = c0 + tid;
        if (c < NCRYS) {
            float s = ob[0];
            for (int f = 0; f < RD; ++f) s += sC[tid][f] * oW[f];
            out[c] = s;
        }
    }
}

// ---------------- launch ----------------
extern "C" void kernel_launch(void* const* d_in, const int* in_sizes, int n_in,
                              void* d_out, int out_size, void* d_ws, size_t ws_size,
                              hipStream_t stream) {
    const float* atom_fea = (const float*)d_in[0];
    const float* nbr_fea  = (const float*)d_in[1];
    const int*   nbr_idx  = (const int*)d_in[2];
    const int*   cbatch   = (const int*)d_in[3];
    const float* atom_W = (const float*)d_in[5];
    const float* atom_b = (const float*)d_in[6];
    const float* edge_W = (const float*)d_in[7];
    const float* edge_b = (const float*)d_in[8];
    const float* eW1 = (const float*)d_in[9];
    const float* eb1 = (const float*)d_in[10];
    const float* eW2 = (const float*)d_in[11];
    const float* eb2 = (const float*)d_in[12];
    const float* nW1 = (const float*)d_in[13];
    const float* nb1 = (const float*)d_in[14];
    const float* nW2 = (const float*)d_in[15];
    const float* nb2 = (const float*)d_in[16];
    const float* read_W = (const float*)d_in[17];
    const float* read_b = (const float*)d_in[18];
    const float* out_W  = (const float*)d_in[19];
    const float* out_b  = (const float*)d_in[20];
    float* out = (float*)d_out;

    // ---- workspace carve (256B aligned) ----
    char* p = (char*)d_ws;
    auto take = [&](size_t bytes) -> void* {
        void* r = (void*)p;
        p += (bytes + 255) & ~(size_t)255;
        return r;
    };
    float* node  = (float*)take((size_t)NNODE * ND * 4);   // 25.6 MB
    float* edge  = (float*)take((size_t)NEDGE * ED * 4);   // 153.6 MB
    float* maskb = (float*)take((size_t)NEDGE * 4);
    float* aggS  = (float*)take((size_t)NNODE * ED * 4);
    float* aggC  = (float*)take((size_t)NNODE * 4);
    float* poolS = (float*)take((size_t)NCRYS * ND * 4);
    float* poolC = (float*)take((size_t)NCRYS * 4);
    // packed bf16 weights (elem counts)
    unsigned short* pAtomW = (unsigned short*)take((size_t)3 * 8 * 512 * 2);
    unsigned short* pEdgeW = (unsigned short*)take((size_t)2 * 4 * 512 * 2);
    unsigned short* pEW1   = (unsigned short*)take((size_t)NLAYER * 10 * 8 * 512 * 2);
    unsigned short* pEW2   = (unsigned short*)take((size_t)NLAYER * 4 * 4 * 512 * 2);
    unsigned short* pNW1   = (unsigned short*)take((size_t)NLAYER * 6 * 8 * 512 * 2);
    unsigned short* pNW2   = (unsigned short*)take((size_t)NLAYER * 4 * 8 * 512 * 2);
    unsigned short* pReadW = (unsigned short*)take((size_t)4 * 8 * 512 * 2);

    const int dsEW1 = 10 * 8 * 512, dsEW2 = 4 * 4 * 512;
    const int dsNW1 = 6 * 8 * 512,  dsNW2 = 4 * 8 * 512;

    // ---- prepack weights to WMMA fragment layout ----
    k_prepack<<<dim3(3, 8, 1), 32, 0, stream>>>(atom_W, pAtomW, AF, ND, 0, 0);
    k_prepack<<<dim3(2, 4, 1), 32, 0, stream>>>(edge_W, pEdgeW, EF, ED, 0, 0);
    k_prepack<<<dim3(10, 8, NLAYER), 32, 0, stream>>>(eW1, pEW1, 320, HID, 320 * HID, dsEW1);
    k_prepack<<<dim3(4, 4, NLAYER), 32, 0, stream>>>(eW2, pEW2, HID, ED, HID * ED, dsEW2);
    k_prepack<<<dim3(6, 8, NLAYER), 32, 0, stream>>>(nW1, pNW1, 192, ND, 192 * ND, dsNW1);
    k_prepack<<<dim3(4, 8, NLAYER), 32, 0, stream>>>(nW2, pNW2, ND, ND, ND * ND, dsNW2);
    k_prepack<<<dim3(4, 8, 1), 32, 0, stream>>>(read_W, pReadW, ND, RD, 0, 0);

    // ---- input projections ----
    k_node_proj<<<NNODE / 16, 256, 0, stream>>>(atom_fea, pAtomW, atom_b, node);
    k_edge_proj<<<NEDGE / 16, 128, 0, stream>>>(nbr_fea, pEdgeW, edge_b, edge, maskb);

    // ---- message-passing layers ----
    for (int l = 0; l < NLAYER; ++l) {
        k_zero<<<(NNODE * ED + 255) / 256, 256, 0, stream>>>(aggS, NNODE * ED);
        k_zero<<<(NNODE + 255) / 256, 256, 0, stream>>>(aggC, NNODE);
        k_edge_update<<<NEDGE / 32, 256, 0, stream>>>(
            node, edge, maskb, nbr_idx,
            pEW1 + (size_t)l * dsEW1, eb1 + (size_t)l * HID,
            pEW2 + (size_t)l * dsEW2, eb2 + (size_t)l * ED,
            aggS, aggC);
        k_node_update<<<NNODE / 16, 256, 0, stream>>>(
            node, aggS, aggC,
            pNW1 + (size_t)l * dsNW1, nb1 + (size_t)l * ND,
            pNW2 + (size_t)l * dsNW2, nb2 + (size_t)l * ND);
    }

    // ---- pooling + readout ----
    k_zero<<<(NCRYS * ND + 255) / 256, 256, 0, stream>>>(poolS, NCRYS * ND);
    k_zero<<<(NCRYS + 255) / 256, 256, 0, stream>>>(poolC, NCRYS);
    k_pool<<<(int)(((long)NNODE * ND + 255) / 256), 256, 0, stream>>>(node, cbatch, poolS, poolC);
    k_readout<<<(NCRYS + 15) / 16, 256, 0, stream>>>(poolS, poolC, pReadW, read_b,
                                                     out_W, out_b, out);
}
